// ContentAttention_56659208568962
// MI455X (gfx1250) — matrix operand order repaired and checked by reading, not compiled, not run on hardware
//
#include <hip/hip_runtime.h>

typedef __attribute__((ext_vector_type(16))) __bf16 v16bf;
typedef __attribute__((ext_vector_type(8)))  float  v8f;

#define B_ 64
#define T_ 2048
#define D_ 256
#define NTILE_T (T_ / 16)      // 128 T-tiles per batch
#define TSPLIT 32
#define EPS_ 1e-7f

// ---------------------------------------------------------------------------
// Kernel 0: convert context_w (fp32, row-major [k][n]) to bf16, transposed to
// column-major [n][k], so WMMA B-fragments become contiguous 32-byte loads.
// ---------------------------------------------------------------------------
__global__ void ca_wprep(const float* __restrict__ ctx_w,
                         __bf16* __restrict__ wbT) {
  const int k = blockIdx.x;    // row of ctx_w
  const int n = threadIdx.x;   // column of ctx_w
  wbT[(size_t)n * D_ + k] = (__bf16)ctx_w[(size_t)k * D_ + n];  // hw RNE cvt
}

// ---------------------------------------------------------------------------
// Kernel 1: bias[b,d] = aspect[b,:] @ aspect_w[:,d] + sentence[b,:] @ sent_w[:,d]
// ---------------------------------------------------------------------------
__global__ void ca_bias(const float* __restrict__ aspect,
                        const float* __restrict__ sentence,
                        const float* __restrict__ asp_w,
                        const float* __restrict__ sen_w,
                        float* __restrict__ bias) {
  const int b = blockIdx.x;
  const int d = threadIdx.x;
  float acc = 0.f;
  for (int k = 0; k < D_; ++k) {
    acc += aspect[b * D_ + k]   * asp_w[k * D_ + d];
    acc += sentence[b * D_ + k] * sen_w[k * D_ + d];
  }
  bias[b * D_ + d] = acc;
}

// ---------------------------------------------------------------------------
// Kernel 2 (hot): fused GEMM(bf16 WMMA) + tanh + dot(attend_w) + exp*mask.
// Block = 128 threads = 4 waves; one 16-row T-tile of one batch per block.
// Wave w handles N-tiles [4w .. 4w+3] of 16 cols each (D = 256 = 16 tiles).
// ---------------------------------------------------------------------------
__global__ void ca_pass1(const float* __restrict__ context,
                         const __bf16* __restrict__ wbT,   // [n][k] bf16
                         const float* __restrict__ bias,
                         const float* __restrict__ attend_w,
                         const int*   __restrict__ cmask,
                         float* __restrict__ e_out,
                         float* __restrict__ tilesum) {
  const int tid  = threadIdx.x;
  const int wave = tid >> 5;
  const int lane = tid & 31;
  const int half = lane >> 4;   // K-half within WMMA A/B fragments
  const int lrow = lane & 15;   // M (for A) / N (for B,C) index

  const int b    = blockIdx.x >> 7;      // / NTILE_T
  const int tile = blockIdx.x & (NTILE_T - 1);
  const int t0   = tile << 4;

  v8f acc[4] = {};   // 4 x 16x16 f32 accumulators

  const float* arow = context + ((size_t)b * T_ + t0 + lrow) * D_;
  const int ncolbase = wave * 64 + lrow;

  for (int kk = 0; kk < 8; ++kk) {            // K = 256, steps of 32
    const int k0 = kk * 32;

    // ---- A fragment: 16x32 bf16. lane(0..15): M=lane, K 0-15;
    //      lane(16..31): M=lane-16, K 16-31. Elements 2r,2r+1 = VGPR r.
    //      8 consecutive fp32 per group -> two b128 loads, hw cvt to bf16.
    v16bf a;
    {
      const float4* p0 = (const float4*)(arow + k0 + half * 16);
      const float4* p1 = (const float4*)(arow + k0 + half * 16 + 8);
      const float4 x0 = p0[0], x1 = p0[1];
      const float4 y0 = p1[0], y1 = p1[1];
      a[0]  = (__bf16)x0.x; a[1]  = (__bf16)x0.y;
      a[2]  = (__bf16)x0.z; a[3]  = (__bf16)x0.w;
      a[4]  = (__bf16)x1.x; a[5]  = (__bf16)x1.y;
      a[6]  = (__bf16)x1.z; a[7]  = (__bf16)x1.w;
      a[8]  = (__bf16)y0.x; a[9]  = (__bf16)y0.y;
      a[10] = (__bf16)y0.z; a[11] = (__bf16)y0.w;
      a[12] = (__bf16)y1.x; a[13] = (__bf16)y1.y;
      a[14] = (__bf16)y1.z; a[15] = (__bf16)y1.w;
    }

    // ---- B fragments (one per N-tile): 32x16 bf16. Lane holds column ncol,
    //      K = k0 + half*16 .. +15 : contiguous 32 bytes in transposed wbT.
#pragma unroll
    for (int j = 0; j < 4; ++j) {
      const int ncol = ncolbase + j * 16;
      const v16bf bb =
          *(const v16bf*)(wbT + (size_t)ncol * D_ + k0 + half * 16);
      acc[j] = __builtin_amdgcn_wmma_f32_16x16x32_bf16(
          false, a, false, bb, (short)0, acc[j], false, false);
    }
  }

  // ---- Epilogue: g[m] = sum_n tanh(acc + bias[n]) * attend_w[n]
  // C layout: VGPR r, lanes 0-15 -> (M=r, N=lane); lanes 16-31 -> (M=r+8).
  float p[8];
#pragma unroll
  for (int r = 0; r < 8; ++r) p[r] = 0.f;

#pragma unroll
  for (int j = 0; j < 4; ++j) {
    const int ncol = ncolbase + j * 16;
    const float bv = bias[b * D_ + ncol];
    const float aw = attend_w[ncol];
#pragma unroll
    for (int r = 0; r < 8; ++r) p[r] += tanhf(acc[j][r] + bv) * aw;
  }

  // reduce over the 16 lanes of each half (i.e., over N)
#pragma unroll
  for (int r = 0; r < 8; ++r) {
#pragma unroll
    for (int m = 1; m < 16; m <<= 1) p[r] += __shfl_xor(p[r], m, 16);
  }

  __shared__ float lds_g[4][16];
  if (lrow == 0) {
#pragma unroll
    for (int r = 0; r < 8; ++r) lds_g[wave][r + half * 8] = p[r];
  }
  __syncthreads();

  if (tid < 16) {
    const float g = lds_g[0][tid] + lds_g[1][tid] + lds_g[2][tid] + lds_g[3][tid];
    const int t = t0 + tid;
    const float mk = (cmask[(size_t)b * T_ + t] != 0) ? 1.0f : 0.0f;
    const float e = expf(g) * mk;   // faithful: no max-subtract
    e_out[(size_t)b * T_ + t] = e;
    float s = e;
#pragma unroll
    for (int m = 1; m < 16; m <<= 1) s += __shfl_xor(s, m, 16);
    if (tid == 0) tilesum[b * NTILE_T + tile] = s;
  }
}

// ---------------------------------------------------------------------------
// Kernel 3: deterministic reduce of 128 tile sums -> sums[b]
// ---------------------------------------------------------------------------
__global__ void ca_redsum(const float* __restrict__ tilesum,
                          float* __restrict__ sums) {
  __shared__ float sm[NTILE_T];
  const int b = blockIdx.x, tid = threadIdx.x;
  sm[tid] = tilesum[b * NTILE_T + tid];
  __syncthreads();
  for (int s = NTILE_T / 2; s > 0; s >>= 1) {
    if (tid < s) sm[tid] += sm[tid + s];
    __syncthreads();
  }
  if (tid == 0) sums[b] = sm[0];
}

// ---------------------------------------------------------------------------
// Kernel 4: split-T weighted sum partials: part[b,s,d] = sum_t context*e
// ---------------------------------------------------------------------------
__global__ void ca_pass2a(const float* __restrict__ context,
                          const float* __restrict__ e_buf,
                          float* __restrict__ part) {
  const int s = blockIdx.x & (TSPLIT - 1);
  const int b = blockIdx.x / TSPLIT;
  const int d = threadIdx.x;
  const int tbeg = s * (T_ / TSPLIT);
  const float* cp = context + ((size_t)b * T_ + tbeg) * D_ + d;
  const float* ep = e_buf + (size_t)b * T_ + tbeg;
  float acc = 0.f;
  for (int i = 0; i < T_ / TSPLIT; ++i) acc += cp[(size_t)i * D_] * ep[i];
  part[((size_t)b * TSPLIT + s) * D_ + d] = acc;
}

// ---------------------------------------------------------------------------
// Kernel 5: out[b,d] = (sum_s part) / (sums[b]+EPS) + sentence[b,d]
// ---------------------------------------------------------------------------
__global__ void ca_pass2b(const float* __restrict__ part,
                          const float* __restrict__ sums,
                          const float* __restrict__ sentence,
                          float* __restrict__ out) {
  const int b = blockIdx.x, d = threadIdx.x;
  const float scale = 1.0f / (sums[b] + EPS_);
  float acc = 0.f;
#pragma unroll
  for (int s = 0; s < TSPLIT; ++s) acc += part[((size_t)b * TSPLIT + s) * D_ + d];
  out[b * D_ + d] = acc * scale + sentence[b * D_ + d];
}

// ---------------------------------------------------------------------------
extern "C" void kernel_launch(void* const* d_in, const int* in_sizes, int n_in,
                              void* d_out, int out_size, void* d_ws, size_t ws_size,
                              hipStream_t stream) {
  const float* context  = (const float*)d_in[0];
  const float* aspect   = (const float*)d_in[1];
  const float* sentence = (const float*)d_in[2];
  const int*   cmask    = (const int*)d_in[3];
  const float* ctx_w    = (const float*)d_in[4];
  const float* asp_w    = (const float*)d_in[5];
  const float* sen_w    = (const float*)d_in[6];
  const float* att_w    = (const float*)d_in[7];
  float* out = (float*)d_out;

  // workspace layout
  float* ws      = (float*)d_ws;
  float* bias    = ws;                          // B*D        = 16384 f
  float* e_buf   = bias + B_ * D_;              // B*T        = 131072 f
  float* tilesum = e_buf + (size_t)B_ * T_;     // B*128      = 8192 f
  float* sums    = tilesum + B_ * NTILE_T;      // B          = 64 f
  float* part    = sums + B_;                   // B*32*D     = 524288 f
  __bf16* wbT    = (__bf16*)(part + (size_t)B_ * TSPLIT * D_);  // D*D bf16

  ca_wprep <<<D_,            D_,  0, stream>>>(ctx_w, wbT);
  ca_bias  <<<B_,            D_,  0, stream>>>(aspect, sentence, asp_w, sen_w, bias);
  ca_pass1 <<<B_ * NTILE_T,  128, 0, stream>>>(context, wbT, bias, att_w, cmask,
                                               e_buf, tilesum);
  ca_redsum<<<B_,            NTILE_T, 0, stream>>>(tilesum, sums);
  ca_pass2a<<<B_ * TSPLIT,   D_,  0, stream>>>(context, e_buf, part);
  ca_pass2b<<<B_,            D_,  0, stream>>>(part, sums, sentence, out);
}
